// PSNMSG_76716705841954
// MI455X (gfx1250) — matrix operand order repaired and checked by reading, not compiled, not run on hardware
//
#include <hip/hip_runtime.h>
#include <hip/hip_bf16.h>

typedef __attribute__((ext_vector_type(16))) _Float16 v16h;
typedef __attribute__((ext_vector_type(8)))  float    v8f;

#define BN_EPS 1e-5f

// ---------------------------------------------------------------------------
// WMMA fragment loaders for V_WMMA_F32_16X16X32_F16 (wave32).
// A (16xK=32, f16): lane m=lane&15, hi=lane>>4; VGPR v<4 -> K=2v(+8*hi),
//                   v>=4 -> K=2v+8(+8*hi)  (per ISA 7.12.2 16-bit A table)
// B (K=32 x 16, f16): lane n=n0+(lane&15); lanes0-15 K=0..7,16..23, lanes16-31 K=8..15,24..31 (same order as A)
// C/D (16x16 f32): VGPR r holds row m=r+8*(lane>>4), col n=lane&15
// ---------------------------------------------------------------------------
__device__ inline v16h frag_A(const _Float16* base, int stride, int k0, int lane) {
  const int m = lane & 15, hi = lane >> 4;
  const _Float16* r = base + m * stride + k0 + 8 * hi;
  v16h a;
#pragma unroll
  for (int v = 0; v < 8; ++v) {
    const int k = (v < 4) ? (2 * v) : (2 * v + 8);
    a[2 * v]     = r[k];
    a[2 * v + 1] = r[k + 1];
  }
  return a;
}

__device__ inline v16h frag_B(const _Float16* wp, int cin, int k0, int n0, int lane) {
  const int n = n0 + (lane & 15), hi = lane >> 4;
  const _Float16* r = wp + n * cin + k0 + 8 * hi;
  v16h b;
#pragma unroll
  for (int v = 0; v < 16; ++v) b[v] = r[(v < 8) ? v : (v + 8)];
  return b;
}

__device__ inline void bn_relu_store(v8f c, const float* bnA, const float* bnB,
                                     int chOff, int n0, _Float16* dst, int stride, int lane) {
  const int nl = lane & 15, hi = lane >> 4;
  const int ch = n0 + nl;
  const float A  = bnA[chOff + ch];
  const float Bc = bnB[chOff + ch];
#pragma unroll
  for (int r = 0; r < 8; ++r) {
    float y = fmaxf(0.f, c[r] * A + Bc);
    dst[(r + 8 * hi) * stride + ch] = (_Float16)y;
  }
}

// ---------------------------------------------------------------------------
// Kernel 1: fused MLP (3->32->64->256, BN+ReLU) + scoring layer 256->512.
// Writes LOGITS transposed: Qt[b][s][m] (sigmoid is monotonic -> same top-k).
// Block = 256 threads = 8 waves, each wave owns 16 points. All matmuls via
// v_wmma_f32_16x16x32_f16 with f16 weights staged in LDS.
// ---------------------------------------------------------------------------
__global__ __launch_bounds__(256) void fused_mlp_score(
    const float* __restrict__ coord,
    const float* __restrict__ w0, const float* __restrict__ b0, const float* __restrict__ g0,
    const float* __restrict__ be0, const float* __restrict__ mu0, const float* __restrict__ var0,
    const float* __restrict__ w1, const float* __restrict__ b1, const float* __restrict__ g1,
    const float* __restrict__ be1, const float* __restrict__ mu1, const float* __restrict__ var1,
    const float* __restrict__ w2, const float* __restrict__ b2, const float* __restrict__ g2,
    const float* __restrict__ be2, const float* __restrict__ mu2, const float* __restrict__ var2,
    const float* __restrict__ w3, const float* __restrict__ b3,
    float* __restrict__ Qt)
{
  __shared__ _Float16 w0p[32 * 32];      //  2 KB (cin padded 3->32 with zeros)
  __shared__ _Float16 w1p[64 * 32];      //  4 KB
  __shared__ _Float16 w2p[256 * 64];     // 32 KB
  __shared__ _Float16 w3c[64 * 256];     // 32 KB (current 64-row s-chunk of w3)
  __shared__ float bnA[352], bnB[352];   // folded BN affine per channel
  __shared__ alignas(16) _Float16 bufS[8][16 * 72];   // per-wave ping (and f32 stage)
  __shared__ alignas(16) _Float16 bufL[8][16 * 264];  // per-wave pong (max 256 ch)

  const int tid  = threadIdx.x;
  const int lane = tid & 31, wv = tid >> 5;
  const int bIdx = blockIdx.x >> 7;            // batch (128 blocks per batch)
  const int mm   = (blockIdx.x & 127) * 128 + wv * 16;  // m base of this wave

  // --- stage weights f32 -> f16, precompute BN affine: y = h*A + B ---------
  for (int i = tid; i < 32 * 32; i += 256) {
    int r = i >> 5, k = i & 31;
    w0p[i] = (k < 3) ? (_Float16)w0[r * 3 + k] : (_Float16)0.f;
  }
  for (int i = tid; i < 64 * 32;  i += 256) w1p[i] = (_Float16)w1[i];
  for (int i = tid; i < 256 * 64; i += 256) w2p[i] = (_Float16)w2[i];
  for (int i = tid; i < 352; i += 256) {
    const float *bb, *gg, *bebe, *mumu, *vv; int lo;
    if (i < 32)      { bb = b0; gg = g0; bebe = be0; mumu = mu0; vv = var0; lo = i; }
    else if (i < 96) { bb = b1; gg = g1; bebe = be1; mumu = mu1; vv = var1; lo = i - 32; }
    else             { bb = b2; gg = g2; bebe = be2; mumu = mu2; vv = var2; lo = i - 96; }
    float s = gg[lo] * rsqrtf(vv[lo] + BN_EPS);
    bnA[i] = s;
    bnB[i] = (bb[lo] - mumu[lo]) * s + bebe[lo];
  }
  __syncthreads();

  _Float16* bS  = bufS[wv];
  _Float16* bL  = bufL[wv];
  float*    stg = (float*)bS;   // 16x17 f32 logit staging (reused during L3)

  // --- load coordinates into bufS as 16x32 f16 (K zero padded) -------------
  {
    const int m = lane & 15, j0 = (lane >> 4) * 16;
    const long long g = ((long long)bIdx * 16384 + mm + m) * 3;
#pragma unroll
    for (int j = 0; j < 16; ++j) {
      int col = j0 + j;
      bS[m * 72 + col] = (col < 3) ? (_Float16)coord[g + col] : (_Float16)0.f;
    }
  }
  __builtin_amdgcn_wave_barrier();

  // --- layer0: 3(pad32) -> 32 ----------------------------------------------
  {
    v16h a = frag_A(bS, 72, 0, lane);
#pragma unroll
    for (int t = 0; t < 2; ++t) {
      v8f c = {};
      v16h b = frag_B(w0p, 32, 0, t * 16, lane);
      c = __builtin_amdgcn_wmma_f32_16x16x32_f16(false, a, false, b, (short)0, c, false, false);
      bn_relu_store(c, bnA, bnB, 0, t * 16, bL, 264, lane);
    }
  }
  __builtin_amdgcn_wave_barrier();

  // --- layer1: 32 -> 64 ----------------------------------------------------
  {
    v16h a = frag_A(bL, 264, 0, lane);
#pragma unroll
    for (int t = 0; t < 4; ++t) {
      v8f c = {};
      v16h b = frag_B(w1p, 32, 0, t * 16, lane);
      c = __builtin_amdgcn_wmma_f32_16x16x32_f16(false, a, false, b, (short)0, c, false, false);
      bn_relu_store(c, bnA, bnB, 32, t * 16, bS, 72, lane);
    }
  }
  __builtin_amdgcn_wave_barrier();

  // --- layer2: 64 -> 256 ---------------------------------------------------
  {
    v16h a0 = frag_A(bS, 72, 0,  lane);
    v16h a1 = frag_A(bS, 72, 32, lane);
#pragma unroll
    for (int t = 0; t < 16; ++t) {
      v8f c = {};
      v16h bb0 = frag_B(w2p, 64, 0,  t * 16, lane);
      c = __builtin_amdgcn_wmma_f32_16x16x32_f16(false, a0, false, bb0, (short)0, c, false, false);
      v16h bb1 = frag_B(w2p, 64, 32, t * 16, lane);
      c = __builtin_amdgcn_wmma_f32_16x16x32_f16(false, a1, false, bb1, (short)0, c, false, false);
      bn_relu_store(c, bnA, bnB, 96, t * 16, bL, 264, lane);
    }
  }
  __builtin_amdgcn_wave_barrier();

  // --- layer3: 256 -> 512 logits, chunked by 64 s-rows ---------------------
  for (int c0 = 0; c0 < 512; c0 += 64) {
    __syncthreads();                       // previous chunk fully consumed
    for (int i = tid; i < 64 * 256; i += 256)
      w3c[i] = (_Float16)w3[(long long)(c0 + (i >> 8)) * 256 + (i & 255)];
    __syncthreads();

#pragma unroll 1
    for (int t = 0; t < 4; ++t) {
      v8f c = {};
#pragma unroll
      for (int kk = 0; kk < 8; ++kk) {
        v16h a = frag_A(bL, 264, kk * 32, lane);
        v16h b = frag_B(w3c, 256, kk * 32, t * 16, lane);
        c = __builtin_amdgcn_wmma_f32_16x16x32_f16(false, a, false, b, (short)0, c, false, false);
      }
      const int sBase = c0 + t * 16;
      const int nl = lane & 15, hi = lane >> 4;
      const float bias = b3[sBase + nl];
      __builtin_amdgcn_wave_barrier();
#pragma unroll
      for (int r = 0; r < 8; ++r) stg[nl * 17 + r + 8 * hi] = c[r] + bias;
      __builtin_amdgcn_wave_barrier();
      // transpose-stage -> 64B-contiguous global runs: 4 s-rows per pass
#pragma unroll
      for (int u = 0; u < 4; ++u) {
        const int srow = u * 4 + (lane >> 3);
        const int mi   = (lane & 7) * 2;
        float2 v;
        v.x = stg[srow * 17 + mi];
        v.y = stg[srow * 17 + mi + 1];
        *(float2*)(Qt + ((long long)(bIdx * 512 + sBase + srow) * 16384 + mm + mi)) = v;
      }
      __builtin_amdgcn_wave_barrier();
    }
  }
}

// ---------------------------------------------------------------------------
// Kernel 2: sorted top-64 per row of Qt (4096 rows x 16384).
// The 64KB row is streamed to LDS ONCE via gfx1250 async LDS-DMA
// (global_load_async_to_lds_b128, ASYNCcnt), then both the radix-histogram
// pass and the compaction pass run out of LDS (halves HBM traffic vs. two
// global scans, since Qt (268MB) exceeds the 192MB L2).
// Packed key (key<<14)|(16383-m) => descending value, ascending index ties.
// ---------------------------------------------------------------------------
__global__ __launch_bounds__(256) void topk64(const float* __restrict__ Qt,
                                              int* __restrict__ idxOut)
{
  __shared__ alignas(16) float rowLds[16384];   // 64 KB: one Qt row
  __shared__ int hist[4096];                    // 16 KB
  __shared__ unsigned long long cand[1024];     //  8 KB
  __shared__ int cnt, thrBin;

  const int tid = threadIdx.x;
  const float* row = Qt + (long long)blockIdx.x * 16384;

  // --- async DMA: global row -> LDS (each thread moves 16 x 16B) -----------
  {
    const unsigned ldsBase = (unsigned)(unsigned long long)(&rowLds[0]);
#pragma unroll
    for (int it = 0; it < 16; ++it) {
      const int i = tid + it * 256;              // float4 index
      const unsigned ldsAddr = ldsBase + (unsigned)i * 16u;
      const unsigned gOff    = (unsigned)i * 16u;
      asm volatile("global_load_async_to_lds_b128 %0, %1, %2"
                   :: "v"(ldsAddr), "v"(gOff), "s"(row)
                   : "memory");
    }
    asm volatile("s_wait_asynccnt 0x0" ::: "memory");
  }

  for (int i = tid; i < 4096; i += 256) hist[i] = 0;
  if (tid == 0) cnt = 0;
  __syncthreads();   // LDS row + cleared hist visible block-wide

  // --- pass 1: 12-bit radix histogram on order-preserving keys -------------
#pragma unroll 4
  for (int p = 0; p < 64; ++p) {
    float v = rowLds[tid + p * 256];
    unsigned u = __float_as_uint(v);
    unsigned key = (u & 0x80000000u) ? ~u : (u | 0x80000000u);
    atomicAdd(&hist[key >> 20], 1);
  }
  __syncthreads();

  if (tid == 0) {
    int c = 0, b = 4095;
    for (;;) { c += hist[b]; if (c >= 64 || b == 0) break; --b; }
    thrBin = b;
  }
  __syncthreads();

  // --- pass 2: compact candidates >= threshold bin --------------------------
  const unsigned tb = (unsigned)thrBin;
#pragma unroll 4
  for (int p = 0; p < 64; ++p) {
    const int m = tid + p * 256;
    float v = rowLds[m];
    unsigned u = __float_as_uint(v);
    unsigned key = (u & 0x80000000u) ? ~u : (u | 0x80000000u);
    if ((key >> 20) >= tb) {
      int pos = atomicAdd(&cnt, 1);
      if (pos < 1024)
        cand[pos] = ((unsigned long long)key << 14) | (unsigned)(16383 - m);
    }
  }
  __syncthreads();

  int n = cnt; if (n > 1024) n = 1024;
  int N = 64;  while (N < n) N <<= 1;
  for (int i = n + tid; i < N; i += 256) cand[i] = 0ull;
  __syncthreads();

  // --- bitonic sort (descending) of <= 1024 candidates ----------------------
  for (int k = 2; k <= N; k <<= 1) {
    for (int j = k >> 1; j > 0; j >>= 1) {
      for (int i = tid; i < N; i += 256) {
        int ixj = i ^ j;
        if (ixj > i) {
          bool ascBlock = (i & k) != 0;
          unsigned long long a = cand[i], b = cand[ixj];
          if ((a < b) != ascBlock) { cand[i] = b; cand[ixj] = a; }
        }
      }
      __syncthreads();
    }
  }
  if (tid < 64) idxOut[blockIdx.x * 64 + tid] = 16383 - (int)(cand[tid] & 16383u);
}

// ---------------------------------------------------------------------------
// Kernel 3: gather outputs. One block per (b,s); thread t: j=t/4, q=t%4.
// Output layout (floats): [sp | gp32 | gp64 | sf | gf32 | gf64]
// ---------------------------------------------------------------------------
__global__ __launch_bounds__(256) void gather_out(
    const float* __restrict__ coord, const float* __restrict__ feat,
    const int* __restrict__ idx, float* __restrict__ out)
{
  const int bs = blockIdx.x;            // b*512 + s
  const int b  = bs >> 9;
  const int tid = threadIdx.x;
  const int j = tid >> 2, q = tid & 3;
  const int m = idx[bs * 64 + j];
  const long long g = (long long)b * 16384 + m;

  const long long O0 = 0LL;        // sampled_points (8,512,3)
  const long long O1 = 12288LL;    // gp32 (8,512,32,3)
  const long long O2 = 405504LL;   // gp64 (8,512,64,3)
  const long long O3 = 1191936LL;  // sampled_feature (8,512,64)
  const long long O4 = 1454080LL;  // gf32 (8,512,32,64)
  const long long O5 = 9842688LL;  // gf64 (8,512,64,64)

  const float4* f = (const float4*)(feat + g * 64 + q * 16);
  float4 v0 = f[0], v1 = f[1], v2 = f[2], v3 = f[3];

  float4* d64 = (float4*)(out + O5 + ((long long)bs * 64 + j) * 64 + q * 16);
  d64[0] = v0; d64[1] = v1; d64[2] = v2; d64[3] = v3;
  if (j < 32) {
    float4* d32 = (float4*)(out + O4 + ((long long)bs * 32 + j) * 64 + q * 16);
    d32[0] = v0; d32[1] = v1; d32[2] = v2; d32[3] = v3;
  }
  if (j == 0) {
    float4* ds = (float4*)(out + O3 + (long long)bs * 64 + q * 16);
    ds[0] = v0; ds[1] = v1; ds[2] = v2; ds[3] = v3;
  }
  if (q == 0) {
    float px = coord[g * 3], py = coord[g * 3 + 1], pz = coord[g * 3 + 2];
    float* p64 = out + O2 + ((long long)bs * 64 + j) * 3;
    p64[0] = px; p64[1] = py; p64[2] = pz;
    if (j < 32) {
      float* p32 = out + O1 + ((long long)bs * 32 + j) * 3;
      p32[0] = px; p32[1] = py; p32[2] = pz;
    }
    if (j == 0) {
      float* sp = out + O0 + (long long)bs * 3;
      sp[0] = px; sp[1] = py; sp[2] = pz;
    }
  }
}

// ---------------------------------------------------------------------------
extern "C" void kernel_launch(void* const* d_in, const int* in_sizes, int n_in,
                              void* d_out, int out_size, void* d_ws, size_t ws_size,
                              hipStream_t stream) {
  const float* coord = (const float*)d_in[0];
  const float* feat  = (const float*)d_in[1];
  const float* w0  = (const float*)d_in[2];
  const float* b0  = (const float*)d_in[3];
  const float* g0  = (const float*)d_in[4];
  const float* be0 = (const float*)d_in[5];
  const float* mu0 = (const float*)d_in[6];
  const float* v0  = (const float*)d_in[7];
  const float* w1  = (const float*)d_in[8];
  const float* b1  = (const float*)d_in[9];
  const float* g1  = (const float*)d_in[10];
  const float* be1 = (const float*)d_in[11];
  const float* mu1 = (const float*)d_in[12];
  const float* v1  = (const float*)d_in[13];
  const float* w2  = (const float*)d_in[14];
  const float* b2  = (const float*)d_in[15];
  const float* g2  = (const float*)d_in[16];
  const float* be2 = (const float*)d_in[17];
  const float* mu2 = (const float*)d_in[18];
  const float* v2  = (const float*)d_in[19];
  const float* w3  = (const float*)d_in[20];
  const float* b3  = (const float*)d_in[21];

  float* Qt  = (float*)d_ws;                                   // 8*512*16384 f32 = 256 MB
  int*   idx = (int*)((char*)d_ws + (size_t)8 * 512 * 16384 * 4);  // 8*512*64 i32 = 1 MB

  fused_mlp_score<<<1024, 256, 0, stream>>>(coord,
      w0, b0, g0, be0, mu0, v0,
      w1, b1, g1, be1, mu1, v1,
      w2, b2, g2, be2, mu2, v2,
      w3, b3, Qt);
  topk64<<<4096, 256, 0, stream>>>(Qt, idx);
  gather_out<<<4096, 256, 0, stream>>>(coord, feat, idx, (float*)d_out);
}
